// GraphTripleConv2d_2559800508726
// MI455X (gfx1250) — matrix operand order repaired and checked by reading, not compiled, not run on hardware
//
#include <hip/hip_runtime.h>
#include <hip/hip_bf16.h>

#define DEV __device__ __forceinline__

typedef __bf16 bf16_t;
typedef __attribute__((ext_vector_type(16))) __bf16 bf16x16;
typedef __attribute__((ext_vector_type(8)))  float  f32x8;

namespace {
constexpr int O_ = 512, T_ = 1024, NPIX = 256;
constexpr int DIN = 64, DP = 64, HID = 128, DOUT = 128;
constexpr int PW = 18, PADP = PW * PW; // zero-padded 18x18 pixel tile

// ---- workspace byte offsets ----
constexpr size_t OFF_POOLED = 0;                                        // O*HID*NPIX f32
constexpr size_t OFF_COUNTS = OFF_POOLED + (size_t)O_ * HID * NPIX * 4; // O f32
constexpr size_t OFF_OBJB   = OFF_COUNTS + (size_t)O_ * 4;              // O*DIN*NPIX bf16
constexpr size_t OFF_PREDB  = OFF_OBJB + (size_t)O_ * DIN * NPIX * 2;   // T*DP bf16
constexpr size_t OFF_W1A    = OFF_PREDB + (size_t)T_ * DP * 2;          // 128 x 1728 bf16
constexpr size_t OFF_W1B    = OFF_W1A + (size_t)128 * 1728 * 2;         // 384 x 1152 bf16
constexpr size_t OFF_W2A    = OFF_W1B + (size_t)384 * 1152 * 2;         // 128 x 1152 bf16
constexpr size_t OFF_W2B    = OFF_W2A + (size_t)128 * 1152 * 2;         // 128 x 128 bf16
constexpr size_t OFF_WOA    = OFF_W2B + (size_t)128 * 128 * 2;          // 128 x 576 bf16
constexpr size_t OFF_WOB    = OFF_WOA + (size_t)128 * 576 * 2;          // 128 x 128 bf16
} // namespace

// ---------------- prep kernels ----------------

__global__ void k_zero(float* pooled, float* counts, long n) {
  long stride = (long)gridDim.x * blockDim.x;
  for (long i = (long)blockIdx.x * blockDim.x + threadIdx.x; i < n; i += stride)
    pooled[i] = 0.f;
  for (long i = (long)blockIdx.x * blockDim.x + threadIdx.x; i < O_; i += stride)
    counts[i] = 0.f;
}

__global__ void k_cvt(const float* in, bf16_t* out, long n) {
  long stride = (long)gridDim.x * blockDim.x;
  for (long i = (long)blockIdx.x * blockDim.x + threadIdx.x; i < n; i += stride)
    out[i] = (bf16_t)in[i];
}

// w[M][Ci][3][3] f32  ->  out[M][9*Ci] bf16 with k = r*Ci + ci  (r = kh*3+kw)
__global__ void k_cvt_w3(const float* w, bf16_t* out, int M, int Ci) {
  long n = (long)M * Ci * 9;
  long stride = (long)gridDim.x * blockDim.x;
  for (long i = (long)blockIdx.x * blockDim.x + threadIdx.x; i < n; i += stride) {
    int m   = (int)(i / ((long)Ci * 9));
    int rem = (int)(i % ((long)Ci * 9));
    int ci  = rem / 9;
    int r   = rem % 9;
    out[(long)m * (9 * Ci) + (long)r * Ci + ci] = (bf16_t)w[i];
  }
}

__global__ void k_counts(const int* edges, float* counts) {
  int t = blockIdx.x * blockDim.x + threadIdx.x;
  if (t < T_) {
    atomicAdd(&counts[edges[2 * t + 0]], 1.f);
    atomicAdd(&counts[edges[2 * t + 1]], 1.f);
  }
}

// ---------------- WMMA helpers ----------------

// A fragment: weights Wb[M][Kdim] bf16 row-major.
// 16-bit A 16x32 layout: lane l -> row m = mbase + (l&15); half = l>>4;
// VGPR0-3 hold K = half*8 .. +7 (contiguous 16B), VGPR4-7 hold K = 16+half*8 .. +7.
DEV bf16x16 load_a_frag(const bf16_t* Wb, int Kdim, int mbase, int k0, int lane) {
  const bf16_t* p = Wb + (size_t)(mbase + (lane & 15)) * Kdim + k0 + ((lane >> 4) << 3);
  union { bf16x16 v; uint4 q[2]; } u;
  u.q[0] = *(const uint4*)(p);
  u.q[1] = *(const uint4*)(p + 16);
  return u.v;
}

DEV void acc_zero(f32x8 acc[16]) {
#pragma unroll
  for (int i = 0; i < 16; ++i)
    acc[i] = f32x8{0.f, 0.f, 0.f, 0.f, 0.f, 0.f, 0.f, 0.f};
}

// One wave computes D[16 x 256] = W[16 x K] * im2col(Xp)[K x 256].
// Xp is the zero-padded LDS tile [18*18][CIN]; no halo branches needed:
// output pixel (h,w)=(nt,nlo), tap (kh,kw) -> padded pixel (nt+kh)*18 + (nlo+kw).
template <int CIN, int TAPS>
DEV void conv_wave(const bf16_t* Wb, const bf16_t* __restrict__ Xp, f32x8 acc[16],
                   int mbase, int lane) {
  constexpr int SPR = CIN / 32; // K-steps per tap
  const int nlo = lane & 15, half = lane >> 4;
#pragma unroll 1
  for (int r = 0; r < TAPS; ++r) {
    const int kh = (TAPS == 9) ? (r / 3) : 1;
    const int kw = (TAPS == 9) ? (r % 3) : 1;
#pragma unroll 1
    for (int s = 0; s < SPR; ++s) {
      bf16x16 a = load_a_frag(Wb, TAPS * CIN, mbase, (r * SPR + s) * 32, lane);
      const bf16_t* p0 = Xp + (size_t)(kh * PW + kw + nlo) * CIN + s * 32 + (half << 3);
#pragma unroll
      for (int nt = 0; nt < 16; ++nt) {
        union { bf16x16 v; uint4 q[2]; } u;
        const bf16_t* p = p0 + (size_t)nt * (PW * CIN);
        u.q[0] = *(const uint4*)(p);
        u.q[1] = *(const uint4*)(p + 16);
        acc[nt] = __builtin_amdgcn_wmma_f32_16x16x32_bf16(
            false, a, false, u.v, (short)0, acc[nt], false, false);
      }
    }
  }
}

// Per-channel instance norm + leaky relu over the wave's 16x256 tile,
// result -> padded LDS Y[18*18][C] (border pre-zeroed).  C/D layout:
// vgpr j, lane l -> row = j + 8*(l>>4), col = (l&15); tile nt = pixel row.
DEV void inorm_lrelu_to_lds(const f32x8 acc[16], bf16_t* Y, int C, int chbase, int lane) {
  const int half = lane >> 4, nlo = lane & 15;
#pragma unroll
  for (int j = 0; j < 8; ++j) {
    float s = 0.f, ss = 0.f;
#pragma unroll
    for (int nt = 0; nt < 16; ++nt) { float v = acc[nt][j]; s += v; ss += v * v; }
#pragma unroll
    for (int m = 1; m < 16; m <<= 1) { s += __shfl_xor(s, m, 32); ss += __shfl_xor(ss, m, 32); }
    float mean = s * (1.f / 256.f);
    float rs   = rsqrtf(fmaxf(ss * (1.f / 256.f) - mean * mean, 0.f) + 1e-5f);
    int   ch   = chbase + j + (half << 3);
#pragma unroll
    for (int nt = 0; nt < 16; ++nt) {
      float v = (acc[nt][j] - mean) * rs;
      v = v >= 0.f ? v : 0.1f * v;
      Y[(size_t)((nt + 1) * PW + nlo + 1) * C + ch] = (bf16_t)v;
    }
  }
}

// Same, but f32 to global out[C][256] (optionally accumulate).
DEV void inorm_lrelu_to_global(const f32x8 acc[16], float* out, int chbase, int lane, bool accum) {
  const int half = lane >> 4, nlo = lane & 15;
#pragma unroll
  for (int j = 0; j < 8; ++j) {
    float s = 0.f, ss = 0.f;
#pragma unroll
    for (int nt = 0; nt < 16; ++nt) { float v = acc[nt][j]; s += v; ss += v * v; }
#pragma unroll
    for (int m = 1; m < 16; m <<= 1) { s += __shfl_xor(s, m, 32); ss += __shfl_xor(ss, m, 32); }
    float mean = s * (1.f / 256.f);
    float rs   = rsqrtf(fmaxf(ss * (1.f / 256.f) - mean * mean, 0.f) + 1e-5f);
    int   ch   = chbase + j + (half << 3);
#pragma unroll
    for (int nt = 0; nt < 16; ++nt) {
      float v = (acc[nt][j] - mean) * rs;
      v = v >= 0.f ? v : 0.1f * v;
      float* p = out + (size_t)ch * NPIX + (nt << 4) + nlo;
      if (accum) *p += v; else *p = v;
    }
  }
}

// bias add + segment-sum scatter via f32 global atomics into dst[C][256].
DEV void bias_scatter_atomic(const f32x8 acc[16], float* dst, const float* bias,
                             int chbase, int lane) {
  const int half = lane >> 4, nlo = lane & 15;
#pragma unroll
  for (int j = 0; j < 8; ++j) {
    int   ch = chbase + j + (half << 3);
    float b  = bias[ch];
#pragma unroll
    for (int nt = 0; nt < 16; ++nt)
      atomicAdd(dst + (size_t)ch * NPIX + (nt << 4) + nlo, acc[nt][j] + b);
  }
}

// ---------------- stage 1: per-triple ----------------

__global__ __launch_bounds__(256) void k_triple(
    const int* edges, const bf16_t* OBJB, const bf16_t* PREDB,
    const bf16_t* W1A, const bf16_t* W1B, const float* b1b,
    float* pooled, float* out_p) {
  extern __shared__ uint4 smem4[];
  char*   smem = (char*)smem4;
  bf16_t* Xt   = (bf16_t*)smem;                        // [324][192] bf16 (121.5 KB)
  bf16_t* Y1   = (bf16_t*)(smem + (size_t)PADP * 192 * 2); // [324][128] bf16 (81 KB)

  const int t = blockIdx.x;
  const int tid = threadIdx.x, lane = tid & 31, wv = tid >> 5;
  const int si = edges[2 * t + 0], oi = edges[2 * t + 1];

  // Build zero-padded input tile: concat(obj_maps[s], pred bcast, obj_maps[o]).
  // Also zero Y1's border once (interior is fully rewritten below).
  for (int p = tid; p < PADP; p += 256) {
    const int hh = p / PW, ww = p % PW;
    const bool inb = (hh >= 1) && (hh <= 16) && (ww >= 1) && (ww <= 16);
    const int src = (hh - 1) * 16 + (ww - 1);
    if (inb) {
#pragma unroll 4
      for (int c = 0; c < 64; ++c)
        Xt[(size_t)p * 192 + c] = OBJB[((size_t)si * 64 + c) * NPIX + src];
#pragma unroll 4
      for (int c = 0; c < 64; ++c)
        Xt[(size_t)p * 192 + 64 + c] = PREDB[(size_t)t * 64 + c];
#pragma unroll 4
      for (int c = 0; c < 64; ++c)
        Xt[(size_t)p * 192 + 128 + c] = OBJB[((size_t)oi * 64 + c) * NPIX + src];
    } else {
#pragma unroll 4
      for (int c = 0; c < 192; ++c) Xt[(size_t)p * 192 + c] = (bf16_t)0.f;
#pragma unroll 4
      for (int c = 0; c < 128; ++c) Y1[(size_t)p * 128 + c] = (bf16_t)0.f;
    }
  }
  __syncthreads();

  f32x8 acc[16];

  // conv1a 3x3 192->128 + inorm + lrelu  (bias cancels under inorm)
  acc_zero(acc);
  conv_wave<192, 9>(W1A, Xt, acc, wv * 16, lane);
  inorm_lrelu_to_lds(acc, Y1, 128, wv * 16, lane);
  __syncthreads();

  // conv1b 3x3 128->384, split into 3 chunks of 128 output channels
  // chunk 0: new_s -> atomic segment-sum into pooled[s]
  acc_zero(acc);
  conv_wave<128, 9>(W1B, Y1, acc, wv * 16, lane);
  bias_scatter_atomic(acc, pooled + (size_t)si * HID * NPIX, b1b, wv * 16, lane);

  // chunk 1: new_p = lrelu(inorm(.)) -> output (bias cancels)
  acc_zero(acc);
  conv_wave<128, 9>(W1B, Y1, acc, 128 + wv * 16, lane);
  inorm_lrelu_to_global(acc, out_p + (size_t)t * DOUT * NPIX, wv * 16, lane, false);

  // chunk 2: new_o -> atomic segment-sum into pooled[o]
  acc_zero(acc);
  conv_wave<128, 9>(W1B, Y1, acc, 256 + wv * 16, lane);
  bias_scatter_atomic(acc, pooled + (size_t)oi * HID * NPIX, b1b + 256, wv * 16, lane);
}

// ---------------- stage 2: per-object ----------------

__global__ __launch_bounds__(256) void k_object(
    const bf16_t* OBJB, const float* pooled, const float* counts,
    const bf16_t* W2A, const bf16_t* W2B, const bf16_t* WOA, const bf16_t* WOB,
    float* out_obj) {
  extern __shared__ uint4 smem4[];
  char*   smem = (char*)smem4;
  bf16_t* X    = (bf16_t*)smem;                            // [324][128] (reused [324][64])
  bf16_t* Y    = (bf16_t*)(smem + (size_t)PADP * 128 * 2); // [324][128]

  const int o = blockIdx.x;
  const int tid = threadIdx.x, lane = tid & 31, wv = tid >> 5;
  const float inv = 1.f / fmaxf(counts[o], 1.f);

  // pooled / counts -> zero-padded bf16 tile; zero Y border once.
  for (int p = tid; p < PADP; p += 256) {
    const int hh = p / PW, ww = p % PW;
    const bool inb = (hh >= 1) && (hh <= 16) && (ww >= 1) && (ww <= 16);
    const int src = (hh - 1) * 16 + (ww - 1);
    if (inb) {
#pragma unroll 4
      for (int c = 0; c < 128; ++c)
        X[(size_t)p * 128 + c] = (bf16_t)(pooled[((size_t)o * HID + c) * NPIX + src] * inv);
    } else {
#pragma unroll 4
      for (int c = 0; c < 128; ++c) X[(size_t)p * 128 + c] = (bf16_t)0.f;
#pragma unroll 4
      for (int c = 0; c < 128; ++c) Y[(size_t)p * 128 + c] = (bf16_t)0.f;
    }
  }
  __syncthreads();

  f32x8 acc[16];

  // path A: block(w2a 3x3) then block(w2b 1x1) -> out
  acc_zero(acc);
  conv_wave<128, 9>(W2A, X, acc, wv * 16, lane);
  inorm_lrelu_to_lds(acc, Y, 128, wv * 16, lane);
  __syncthreads();

  acc_zero(acc);
  conv_wave<128, 1>(W2B, Y, acc, wv * 16, lane);
  inorm_lrelu_to_global(acc, out_obj + (size_t)o * DOUT * NPIX, wv * 16, lane, false);
  __syncthreads();

  // path B: block(woa 3x3 64->128) then block(wob 1x1) -> accumulate into out
  for (int p = tid; p < PADP; p += 256) {
    const int hh = p / PW, ww = p % PW;
    const bool inb = (hh >= 1) && (hh <= 16) && (ww >= 1) && (ww <= 16);
    const int src = (hh - 1) * 16 + (ww - 1);
#pragma unroll 4
    for (int c = 0; c < 64; ++c)
      X[(size_t)p * 64 + c] = inb ? OBJB[((size_t)o * DIN + c) * NPIX + src] : (bf16_t)0.f;
  }
  __syncthreads();

  acc_zero(acc);
  conv_wave<64, 9>(WOA, X, acc, wv * 16, lane);
  inorm_lrelu_to_lds(acc, Y, 128, wv * 16, lane);
  __syncthreads();

  acc_zero(acc);
  conv_wave<128, 1>(WOB, Y, acc, wv * 16, lane);
  inorm_lrelu_to_global(acc, out_obj + (size_t)o * DOUT * NPIX, wv * 16, lane, true);
}

// ---------------- launcher ----------------

extern "C" void kernel_launch(void* const* d_in, const int* in_sizes, int n_in,
                              void* d_out, int out_size, void* d_ws, size_t ws_size,
                              hipStream_t stream) {
  (void)in_sizes; (void)n_in; (void)out_size; (void)ws_size;

  const float* obj_maps  = (const float*)d_in[0];
  const float* pred_vecs = (const float*)d_in[1];
  const int*   edges     = (const int*)d_in[2];
  const float* w1a       = (const float*)d_in[4];
  const float* w1b       = (const float*)d_in[6];
  const float* b1b       = (const float*)d_in[7];
  const float* w2a       = (const float*)d_in[8];
  const float* w2b       = (const float*)d_in[10];
  const float* woa       = (const float*)d_in[12];
  const float* wob       = (const float*)d_in[14];

  char*   ws     = (char*)d_ws;
  float*  pooled = (float*)(ws + OFF_POOLED);
  float*  counts = (float*)(ws + OFF_COUNTS);
  bf16_t* OBJB   = (bf16_t*)(ws + OFF_OBJB);
  bf16_t* PREDB  = (bf16_t*)(ws + OFF_PREDB);
  bf16_t* W1A    = (bf16_t*)(ws + OFF_W1A);
  bf16_t* W1B    = (bf16_t*)(ws + OFF_W1B);
  bf16_t* W2A    = (bf16_t*)(ws + OFF_W2A);
  bf16_t* W2B    = (bf16_t*)(ws + OFF_W2B);
  bf16_t* WOA    = (bf16_t*)(ws + OFF_WOA);
  bf16_t* WOB    = (bf16_t*)(ws + OFF_WOB);

  float* out_obj = (float*)d_out;                      // (O,128,16,16)
  float* out_p   = out_obj + (size_t)O_ * DOUT * NPIX; // (T,128,16,16)

  const int smem_triple = PADP * 192 * 2 + PADP * 128 * 2; // ~202 KB
  const int smem_object = 2 * PADP * 128 * 2;              // ~162 KB
  hipFuncSetAttribute((const void*)k_triple, hipFuncAttributeMaxDynamicSharedMemorySize, smem_triple);
  hipFuncSetAttribute((const void*)k_object, hipFuncAttributeMaxDynamicSharedMemorySize, smem_object);

  k_zero<<<2048, 256, 0, stream>>>(pooled, counts, (long)O_ * HID * NPIX);
  k_cvt<<<2048, 256, 0, stream>>>(obj_maps, OBJB, (long)O_ * DIN * NPIX);
  k_cvt<<<64, 256, 0, stream>>>(pred_vecs, PREDB, (long)T_ * DP);
  k_cvt_w3<<<256, 256, 0, stream>>>(w1a, W1A, 128, 192);
  k_cvt_w3<<<512, 256, 0, stream>>>(w1b, W1B, 384, 128);
  k_cvt_w3<<<256, 256, 0, stream>>>(w2a, W2A, 128, 128);
  k_cvt<<<64, 256, 0, stream>>>(w2b, W2B, 128 * 128);
  k_cvt_w3<<<128, 256, 0, stream>>>(woa, WOA, 128, 64);
  k_cvt<<<64, 256, 0, stream>>>(wob, WOB, 128 * 128);
  k_counts<<<(T_ + 255) / 256, 256, 0, stream>>>(edges, counts);

  k_triple<<<T_, 256, smem_triple, stream>>>(edges, OBJB, PREDB, W1A, W1B, b1b, pooled, out_p);
  k_object<<<O_, 256, smem_object, stream>>>(OBJB, pooled, counts, W2A, W2B, WOA, WOB, out_obj);
}
